// OutputBlock_42082089566316
// MI455X (gfx1250) — matrix operand order repaired and checked by reading, not compiled, not run on hardware
//
#include <hip/hip_runtime.h>
#include <hip/hip_bf16.h>

// ---------------- problem constants (match reference) ----------------
#define HID        128      // HIDDEN
#define NRAD       6        // NUM_RADIAL
#define OEMB       256      // OUT_EMB
#define NLAYERS    3
#define NNODES     40000
#define TILE_ROWS  128      // node rows per block in the MLP kernel
#define LDA        264      // padded bf16 K-stride for activations (256+8)
#define LDW        264      // padded bf16 K-stride for weights (264*2B = 33*16B)

typedef __attribute__((ext_vector_type(16))) __bf16 v16bf;
typedef __attribute__((ext_vector_type(8)))  float  v8f;
typedef int v4i __attribute__((vector_size(16)));   // matches builtin's V4i

#define AS_GLOBAL __attribute__((address_space(1)))
#define AS_LDS    __attribute__((address_space(3)))

// =====================================================================
// CDNA5 async global -> LDS copy (16 B per lane, tracked by ASYNCcnt).
// Toolchain signature (from diagnostics):
//   void __builtin_amdgcn_global_load_async_to_lds_b128(
//       v4i AS(1)* src, v4i AS(3)* lds_dst, imm int offset, imm int cpol)
// =====================================================================
__device__ __forceinline__ void async_copy_b128(const void* gsrc, void* ldst) {
#if __has_builtin(__builtin_amdgcn_global_load_async_to_lds_b128)
  __builtin_amdgcn_global_load_async_to_lds_b128(
      (AS_GLOBAL v4i*)(unsigned long long)gsrc,
      (AS_LDS v4i*)(unsigned int)(unsigned long long)ldst,
      0, 0);
#else
  const unsigned int lds_off = (unsigned int)(unsigned long long)ldst;
  asm volatile("global_load_async_to_lds_b128 %0, %1, off"
               :
               : "v"(lds_off), "v"((unsigned long long)gsrc)
               : "memory");
#endif
}

__device__ __forceinline__ void wait_async_zero() {
#if __has_builtin(__builtin_amdgcn_s_wait_asynccnt)
  __builtin_amdgcn_s_wait_asynccnt(0);
#else
  asm volatile("s_wait_asynccnt 0" ::: "memory");
#endif
}

// Stage a [OEMB x K] bf16 weight panel (row-major, contiguous in global)
// into padded LDS via async-to-LDS, 8 bf16 elements per op.
template <int K>
__device__ __forceinline__ void stage_weights_async(const __bf16* __restrict__ g,
                                                    __bf16* sW) {
  constexpr int CPR   = K >> 3;        // 16-byte chunks per row
  constexpr int TOTAL = OEMB * CPR;
  for (int i = threadIdx.x; i < TOTAL; i += 256) {
    const int row = i / CPR;
    const int c8  = i - row * CPR;
    async_copy_b128(g + (size_t)row * K + c8 * 8, sW + row * LDW + c8 * 8);
  }
  wait_async_zero();
}

// =====================================================================
// WMMA fragment loaders (wave32, per CDNA5 ISA layouts)
// A 16x32 bf16: lanes 0-15 = rows M, K in {0..7,16..23}; lanes 16-31 same
// rows, K in {8..15,24..31}; VGPR j holds the K-pair (see ISA table).
// B 32x16 bf16: lane n = column N; K order matches A (lanes 0-15 K in {0..7,16..23}, lanes 16-31 K in {8..15,24..31}).
// C/D 16x16 f32: VGPR r: lanes 0-15 -> M=r, lanes 16-31 -> M=8+r; N=lane%16.
// =====================================================================
__device__ __forceinline__ v16bf load_frag_a(const __bf16* sA, int rowBase, int k0) {
  const int lane = threadIdx.x & 31;
  const int hf   = lane >> 4;
  const int m    = lane & 15;
  const __bf16* p = sA + (rowBase + m) * LDA + k0 + hf * 8;
  v16bf a;
#pragma unroll
  for (int j = 0; j < 8; ++j) {
    const int k = ((j >> 2) << 4) + ((j & 3) << 1);   // 0,2,4,6,16,18,20,22
    a[2 * j]     = p[k];
    a[2 * j + 1] = p[k + 1];
  }
  return a;
}

// B[k][n] = W[colBase+n][k] (weights stored [out][in] row-major in LDS)
__device__ __forceinline__ v16bf load_frag_b(const __bf16* sW, int colBase, int k0) {
  const int lane = threadIdx.x & 31;
  const int hf   = lane >> 4;
  const int n    = lane & 15;
  const __bf16* p = sW + (colBase + n) * LDW + k0 + hf * 8;
  v16bf b;
#pragma unroll
  for (int e = 0; e < 16; ++e) b[e] = p[e < 8 ? e : e + 8];
  return b;
}

// One wave computes a 16-row x 256-col strip as 16 WMMA tiles.
// B (and A) fragments are software-pipelined one step ahead so ds_load
// latency overlaps the WMMA of the previous tile (partial dscnt waits).
template <int K>
__device__ __forceinline__ void gemm_strip(const __bf16* sA, const __bf16* sW,
                                           int rowBase, v8f acc[16]) {
  const v8f z = {0.f, 0.f, 0.f, 0.f, 0.f, 0.f, 0.f, 0.f};
#pragma unroll
  for (int tc = 0; tc < 16; ++tc) acc[tc] = z;

  v16bf a = load_frag_a(sA, rowBase, 0);
  v16bf b = load_frag_b(sW, 0, 0);
#pragma unroll
  for (int k0 = 0; k0 < K; k0 += 32) {
    v16bf a_next = a;
    if (k0 + 32 < K) a_next = load_frag_a(sA, rowBase, k0 + 32);
#pragma unroll
    for (int tc = 0; tc < 16; ++tc) {
      v16bf b_next = b;
      if (tc < 15)          b_next = load_frag_b(sW, (tc + 1) * 16, k0);
      else if (k0 + 32 < K) b_next = load_frag_b(sW, 0, k0 + 32);
      acc[tc] = __builtin_amdgcn_wmma_f32_16x16x32_bf16(
          false, a, false, b, (short)0, acc[tc], false, false);
      b = b_next;
    }
    a = a_next;
  }
}

// =====================================================================
// Kernel 1: zero the fp32 node accumulator (N x HID) in scratch
// =====================================================================
__global__ void zero_nodes_kernel(float4* __restrict__ p, int n4) {
  const float4 z = {0.f, 0.f, 0.f, 0.f};
  for (int i = blockIdx.x * blockDim.x + threadIdx.x; i < n4;
       i += gridDim.x * blockDim.x)
    p[i] = z;
}

// =====================================================================
// Kernel 2: fp32 -> bf16 weight conversion (weights then live in L2)
// =====================================================================
__global__ void cvt_bf16_kernel(const float* __restrict__ src,
                                __bf16* __restrict__ dst, int n) {
  for (int i = blockIdx.x * blockDim.x + threadIdx.x; i < n;
       i += gridDim.x * blockDim.x)
    dst[i] = (__bf16)src[i];
}

// =====================================================================
// Kernel 3: edge-wise g = (rbf @ W_rbf^T) * x, atomic scatter into nodes.
// One wave per edge; lane owns 4 hidden channels (float4 load of x).
// The 20 MB accumulator lives in L2 (192 MB), so atomics are L2 RMWs.
// =====================================================================
__global__ __launch_bounds__(256)
void edge_scatter_kernel(const float* __restrict__ x,
                         const float* __restrict__ rbf,
                         const int*   __restrict__ idx,
                         const float* __restrict__ Wr,   // [HID][NRAD]
                         float* __restrict__ hn,         // [N][HID] accum
                         int E) {
  __shared__ float sWr[HID * NRAD];
  for (int t = threadIdx.x; t < HID * NRAD; t += blockDim.x) sWr[t] = Wr[t];
  __syncthreads();

  const int lane = threadIdx.x & 31;
  const int wave = (blockIdx.x * blockDim.x + threadIdx.x) >> 5;
  const int nw   = (gridDim.x * blockDim.x) >> 5;
  const int c0   = lane * 4;

  for (int e = wave; e < E; e += nw) {
    if (e + nw < E)  // prefetch next edge's x row (global_prefetch_b8)
      __builtin_prefetch(x + (size_t)(e + nw) * HID + c0, 0, 1);

    const float* rb = rbf + (size_t)e * NRAD;
    const float r0 = rb[0], r1 = rb[1], r2 = rb[2],
                r3 = rb[3], r4 = rb[4], r5 = rb[5];
    const float4 xv = *(const float4*)(x + (size_t)e * HID + c0);

    float g[4];
#pragma unroll
    for (int u = 0; u < 4; ++u) {
      const float* w = &sWr[(c0 + u) * NRAD];
      g[u] = w[0] * r0 + w[1] * r1 + w[2] * r2 +
             w[3] * r3 + w[4] * r4 + w[5] * r5;
    }
    g[0] *= xv.x; g[1] *= xv.y; g[2] *= xv.z; g[3] *= xv.w;

    float* dst = hn + (size_t)idx[e] * HID + c0;
    atomicAdd(dst + 0, g[0]);
    atomicAdd(dst + 1, g[1]);
    atomicAdd(dst + 2, g[2]);
    atomicAdd(dst + 3, g[3]);
  }
}

// =====================================================================
// Kernel 4: fused node MLP. One block = 128 node rows, 8 waves; each wave
// owns a 16x256 strip (16 v8f accumulators). bf16 activations/weights in
// padded LDS (weights staged via async-to-LDS); fp32 accumulate; SiLU
// fused; final 256->1 dot in fp32 with a shuffle reduction.
// =====================================================================
__global__ __launch_bounds__(256)
void mlp_fused_kernel(const float*  __restrict__ hn,    // [N][HID] fp32
                      const __bf16* __restrict__ wup,   // [OEMB][HID] bf16
                      const float*  __restrict__ bup,   // [OEMB]
                      const __bf16* __restrict__ wl,    // [L][OEMB][OEMB] bf16
                      const float*  __restrict__ bl,    // [L][OEMB]
                      const float*  __restrict__ wout,  // [OEMB]
                      float* __restrict__ out, int N) {
  extern __shared__ char smem[];
  __bf16* sA = (__bf16*)smem;                                      // 128 x LDA
  __bf16* sW = (__bf16*)(smem + TILE_ROWS * LDA * sizeof(__bf16)); // 256 x LDW

  const int tid      = threadIdx.x;
  const int tileBase = blockIdx.x * TILE_ROWS;

  // Stage W_up via async-to-LDS first (ASYNCcnt path, no VGPR traffic) ...
  stage_weights_async<HID>(wup, sW);
  // ... then, once the copies have landed, VALU converts the node-feature tile fp32 -> bf16.
  for (int i = tid; i < TILE_ROWS * HID; i += 256) {
    const int r = i >> 7, c = i & 127;
    const int gr = tileBase + r;
    const float v = (gr < N) ? hn[(size_t)gr * HID + c] : 0.f;
    sA[r * LDA + c] = (__bf16)v;
  }
  __syncthreads();

  const int wave    = tid >> 5;
  const int rowBase = wave * 16;
  const int lane    = tid & 31;
  const int hf      = lane >> 4;
  const int n       = lane & 15;

  v8f acc[16];

  // ---- up-projection: (128xHID) @ W_up^T + b_up (no activation) ----
  gemm_strip<HID>(sA, sW, rowBase, acc);
#pragma unroll
  for (int tc = 0; tc < 16; ++tc) acc[tc] = acc[tc] + bup[tc * 16 + n];

  __syncthreads();  // all waves finished reading sW(W_up)
  // write activations back to LDS (each wave touches only its own rows)
#pragma unroll
  for (int tc = 0; tc < 16; ++tc)
#pragma unroll
    for (int r = 0; r < 8; ++r)
      sA[(rowBase + hf * 8 + r) * LDA + tc * 16 + n] = (__bf16)acc[tc][r];

  // ---- 3 hidden layers: silu(h @ W_l^T + b_l) ----
  for (int l = 0; l < NLAYERS; ++l) {
    stage_weights_async<OEMB>(wl + (size_t)l * OEMB * OEMB, sW);
    __syncthreads();  // sW ready, sA stores visible

    gemm_strip<OEMB>(sA, sW, rowBase, acc);

    const float* b = bl + l * OEMB;
#pragma unroll
    for (int tc = 0; tc < 16; ++tc) {
      const float bias = b[tc * 16 + n];
#pragma unroll
      for (int r = 0; r < 8; ++r) {
        const float v = acc[tc][r] + bias;
        acc[tc][r] = v / (1.f + __expf(-v));  // SiLU
      }
    }
    __syncthreads();  // all waves done reading sA/sW for this layer
    if (l < NLAYERS - 1) {
#pragma unroll
      for (int tc = 0; tc < 16; ++tc)
#pragma unroll
        for (int r = 0; r < 8; ++r)
          sA[(rowBase + hf * 8 + r) * LDA + tc * 16 + n] = (__bf16)acc[tc][r];
    }
  }

  // ---- final projection: out[m] = sum_c act[m][c] * W_out[c] (fp32) ----
  float part[8] = {0, 0, 0, 0, 0, 0, 0, 0};
#pragma unroll
  for (int tc = 0; tc < 16; ++tc) {
    const float w = wout[tc * 16 + n];
#pragma unroll
    for (int r = 0; r < 8; ++r) part[r] += acc[tc][r] * w;
  }
#pragma unroll
  for (int r = 0; r < 8; ++r) {
    float s = part[r];
    s += __shfl_xor(s, 8, 32);   // stays inside the 16-lane half
    s += __shfl_xor(s, 4, 32);
    s += __shfl_xor(s, 2, 32);
    s += __shfl_xor(s, 1, 32);
    const int grow = tileBase + rowBase + hf * 8 + r;
    if (n == 0 && grow < N) out[grow] = s;
  }
}

// =====================================================================
// Host launcher
// =====================================================================
extern "C" void kernel_launch(void* const* d_in, const int* in_sizes, int n_in,
                              void* d_out, int out_size, void* d_ws, size_t ws_size,
                              hipStream_t stream) {
  const float* x      = (const float*)d_in[0];  // (E, HID)
  const float* rbf    = (const float*)d_in[1];  // (E, NRAD)
  const int*   idx    = (const int*)  d_in[2];  // (E,)
  /* d_in[3] = num_nodes scalar (compile-time NNODES) */
  const float* W_rbf  = (const float*)d_in[4];  // (HID, NRAD)
  const float* W_up   = (const float*)d_in[5];  // (OEMB, HID)
  const float* b_up   = (const float*)d_in[6];  // (OEMB,)
  const float* W_lins = (const float*)d_in[7];  // (L, OEMB, OEMB)
  const float* b_lins = (const float*)d_in[8];  // (L, OEMB)
  const float* W_out  = (const float*)d_in[9];  // (1, OEMB)
  float* out = (float*)d_out;

  const int E = in_sizes[2];
  const int N = NNODES;

  // scratch layout (all 16-B aligned)
  char*   ws     = (char*)d_ws;
  float*  hn     = (float*)ws;                               // N*HID fp32 accum
  size_t  off    = (size_t)N * HID * sizeof(float);
  __bf16* wup_bf = (__bf16*)(ws + off);  off += (size_t)OEMB * HID * sizeof(__bf16);
  __bf16* wl_bf  = (__bf16*)(ws + off);

  // 1) zero node accumulator
  const int n4 = N * HID / 4;
  zero_nodes_kernel<<<(n4 + 255) / 256, 256, 0, stream>>>((float4*)hn, n4);

  // 2) weights -> bf16 (once per call; tiny, stays resident in L2)
  cvt_bf16_kernel<<<(OEMB * HID + 255) / 256, 256, 0, stream>>>(W_up, wup_bf, OEMB * HID);
  cvt_bf16_kernel<<<(NLAYERS * OEMB * OEMB + 255) / 256, 256, 0, stream>>>(
      W_lins, wl_bf, NLAYERS * OEMB * OEMB);

  // 3) edge compute + atomic scatter (memory-bound phase)
  edge_scatter_kernel<<<2048, 256, 0, stream>>>(x, rbf, idx, W_rbf, hn, E);

  // 4) fused node MLP (WMMA phase); ~198 KB dynamic LDS (<= 320 KB/WGP)
  const size_t smem_bytes =
      (size_t)TILE_ROWS * LDA * sizeof(__bf16) + (size_t)OEMB * LDW * sizeof(__bf16);
  (void)hipFuncSetAttribute((const void*)mlp_fused_kernel,
                            hipFuncAttributeMaxDynamicSharedMemorySize,
                            (int)smem_bytes);
  mlp_fused_kernel<<<(N + TILE_ROWS - 1) / TILE_ROWS, 256, smem_bytes, stream>>>(
      hn, wup_bf, b_up, wl_bf, b_lins, W_out, out, N);
}
